// SLSTMCell_91147795956279
// MI455X (gfx1250) — matrix operand order repaired and checked by reading, not compiled, not run on hardware
//
#include <hip/hip_runtime.h>
#include <math.h>

#define B_DIM 4096
#define K_DIM 2048
#define D_DIM 2048

typedef __bf16 bf16;
typedef __attribute__((ext_vector_type(16))) __bf16 v16bf;
typedef __attribute__((ext_vector_type(8)))  __bf16 v8bf;
typedef __attribute__((ext_vector_type(4)))  __bf16 v4bf;
typedef __attribute__((ext_vector_type(8)))  float  v8f;

// Block: 256 threads = 8 wave32. Tile: 64 rows (batch) x 64 cols (d), all 4 gates.
// Wave w: gate g = w>>1, n-half wn = w&1. Wave computes 4(M) x 2(N) accum tiles.
// LDS double-buffered (ping-pong): 1 barrier per K-step.
__global__ __launch_bounds__(256) void slstm_fused_kernel(
    const float* __restrict__ x,      // (B, K)
    const float* __restrict__ c_old,  // (B, D)
    const float* __restrict__ n_old,  // (B, D)
    const float* __restrict__ h_old,  // (B, D)
    const float* __restrict__ m_old,  // (B, D)
    const float* __restrict__ W,      // (4D, K) row-major
    const float* __restrict__ bias,   // (4D)
    const float* __restrict__ rvec,   // (4D)
    float* __restrict__ out)          // h,c,n,m each (B,D), concatenated
{
    __shared__ union {
        struct {
            bf16 A[2][64 * 32];      // [buf][m][k]        bf16 x-tile (ping-pong)
            bf16 Bt[2][256 * 32];    // [buf][g*64+n][k]   bf16 W-tile (4 gates)
        } s;                         // 40 KB
        float pre[4 * 64 * 64];      // [gate][m][d] fp32 pre-activations, 64 KB
    } sm;

    const int tid  = threadIdx.x;
    const int lane = tid & 31;
    const int hh   = lane >> 4;      // lane half (0/1)
    const int lr   = lane & 15;

    const int m0 = blockIdx.x * 64;  // batch-row tile base
    const int d0 = blockIdx.y * 64;  // d tile base

    const int wave = tid >> 5;
    const int g    = wave >> 1;      // gate 0..3 (z,i,f,o)
    const int wn   = wave & 1;       // n-half within the 64-wide d tile

    v8f acc[4][2] = {};

    // -------- register staging for global -> LDS (double buffer pipeline) ----
    float4 ar[2];                    // A tile: 64x32 fp32 = 512 float4 / 256 thr
    float4 br[8];                    // B tile: 256x32 fp32 = 2048 float4 / 256 thr

    auto load_tiles = [&](int k0) {
        #pragma unroll
        for (int i = 0; i < 2; ++i) {
            int idx = tid + i * 256;
            int row = idx >> 3, c4 = (idx & 7) * 4;
            ar[i] = *(const float4*)(x + (size_t)(m0 + row) * K_DIM + k0 + c4);
        }
        #pragma unroll
        for (int i = 0; i < 8; ++i) {
            int idx  = tid + i * 256;
            int wrow = idx >> 3, c4 = (idx & 7) * 4;
            int gg = wrow >> 6, nn = wrow & 63;
            br[i] = *(const float4*)(W + (size_t)(gg * D_DIM + d0 + nn) * K_DIM + k0 + c4);
        }
    };
    auto store_tiles = [&](int buf) {   // fp32 -> bf16 (v_cvt_pk_bf16_f32) + ds_store_b64
        #pragma unroll
        for (int i = 0; i < 2; ++i) {
            int idx = tid + i * 256;
            int row = idx >> 3, c4 = (idx & 7) * 4;
            v4bf p; p.x = (bf16)ar[i].x; p.y = (bf16)ar[i].y;
                    p.z = (bf16)ar[i].z; p.w = (bf16)ar[i].w;
            *(v4bf*)(&sm.s.A[buf][row * 32 + c4]) = p;
        }
        #pragma unroll
        for (int i = 0; i < 8; ++i) {
            int idx  = tid + i * 256;
            int wrow = idx >> 3, c4 = (idx & 7) * 4;
            v4bf p; p.x = (bf16)br[i].x; p.y = (bf16)br[i].y;
                    p.z = (bf16)br[i].z; p.w = (bf16)br[i].w;
            *(v4bf*)(&sm.s.Bt[buf][wrow * 32 + c4]) = p;
        }
    };

    load_tiles(0);

    for (int kk = 0; kk < K_DIM / 32; ++kk) {
        const int buf = kk & 1;
        store_tiles(buf);
        __syncthreads();

        if (kk + 1 < K_DIM / 32) {
            load_tiles((kk + 1) * 32);                 // overlap with compute below
            if (kk + 2 < K_DIM / 32) {                 // warm K-step kk+2 (global_prefetch_b8)
                __builtin_prefetch(x + (size_t)(m0 + (tid >> 2)) * K_DIM + (kk + 2) * 32, 0, 1);
                __builtin_prefetch(W + (size_t)((tid >> 3) * 32 + d0) * K_DIM + (kk + 2) * 32, 0, 1);
            }
        }

        // ---------- LDS -> fragments (contiguous k order, same for A and B) ----------
        // A 16x32 bf16: lane half 0 holds K[0..15], half 1 holds K[16..31]
        union { v16bf v; v8bf h8[2]; } a[4];
        #pragma unroll
        for (int mi = 0; mi < 4; ++mi) {
            const bf16* rowp = &sm.s.A[buf][(mi * 16 + lr) * 32];
            a[mi].v = *(const v16bf*)(rowp + hh * 16);
        }
        // B 32x16 bf16: column = lane%16, half 0 holds K[0..15], half 1 holds K[16..31]
        v16bf b[2];
        #pragma unroll
        for (int ni = 0; ni < 2; ++ni) {
            const bf16* colp = &sm.s.Bt[buf][(g * 64 + wn * 32 + ni * 16 + lr) * 32];
            b[ni] = *(const v16bf*)(colp + hh * 16);
        }

        // ---------- 8 WMMAs per wave per K-step ----------
        #pragma unroll
        for (int mi = 0; mi < 4; ++mi)
            #pragma unroll
            for (int ni = 0; ni < 2; ++ni)
                acc[mi][ni] = __builtin_amdgcn_wmma_f32_16x16x32_bf16(
                    false, a[mi].v, false, b[ni],
                    (short)0, acc[mi][ni], false, false);
    }
    __syncthreads();   // all waves done reading bf16 tiles before union reuse

    // ---------- accumulators -> LDS pre[gate][m][d] ----------
    // C/D layout: VGPR v, lanes 0-15: M=v, N=lane; lanes 16-31: M=8+v, N=lane-16
    #pragma unroll
    for (int mi = 0; mi < 4; ++mi)
        #pragma unroll
        for (int ni = 0; ni < 2; ++ni)
            #pragma unroll
            for (int v = 0; v < 8; ++v) {
                int mm = mi * 16 + v + hh * 8;
                int dd = wn * 32 + ni * 16 + lr;
                sm.pre[(g * 64 + mm) * 64 + dd] = acc[mi][ni][v];
            }
    __syncthreads();

    // ---------- fused sLSTM epilogue ----------
    const size_t BD = (size_t)B_DIM * D_DIM;
    #pragma unroll 4
    for (int i = 0; i < 16; ++i) {
        int pid = tid + i * 256;
        int mm = pid >> 6, dd = pid & 63;       // dd constant per thread -> coalesced on d
        int gm = m0 + mm, gd = d0 + dd;
        size_t gidx = (size_t)gm * D_DIM + gd;

        float hv = h_old[gidx];
        float zp = sm.pre[(0 * 64 + mm) * 64 + dd] + bias[0 * D_DIM + gd] + rvec[0 * D_DIM + gd] * hv;
        float ip = sm.pre[(1 * 64 + mm) * 64 + dd] + bias[1 * D_DIM + gd] + rvec[1 * D_DIM + gd] * hv;
        float fp = sm.pre[(2 * 64 + mm) * 64 + dd] + bias[2 * D_DIM + gd] + rvec[2 * D_DIM + gd] * hv;
        float op = sm.pre[(3 * 64 + mm) * 64 + dd] + bias[3 * D_DIM + gd] + rvec[3 * D_DIM + gd] * hv;

        float mo   = m_old[gidx];
        float zt   = tanhf(zp);
        float ot   = 1.0f / (1.0f + __expf(-op));
        float mnew = fmaxf(fp + mo, ip);        // log-space stabilizer
        float ft   = __expf(fp + mo - mnew);
        float it   = __expf(ip - mnew);
        float cnew = ft * c_old[gidx] + it * zt;
        float nnew = ft * n_old[gidx] + it;
        float hnew = ot * (cnew / (fabsf(nnew) + 1e-6f));

        out[gidx]          = hnew;
        out[BD + gidx]     = cnew;
        out[2 * BD + gidx] = nnew;
        out[3 * BD + gidx] = mnew;
    }
}

extern "C" void kernel_launch(void* const* d_in, const int* in_sizes, int n_in,
                              void* d_out, int out_size, void* d_ws, size_t ws_size,
                              hipStream_t stream) {
    (void)in_sizes; (void)n_in; (void)out_size; (void)d_ws; (void)ws_size;
    const float* x    = (const float*)d_in[0];  // x_t
    const float* c    = (const float*)d_in[1];  // c
    const float* n    = (const float*)d_in[2];  // n
    const float* h    = (const float*)d_in[3];  // h
    const float* m    = (const float*)d_in[4];  // m
    const float* W    = (const float*)d_in[5];  // W_w (4D, IN)
    const float* bias = (const float*)d_in[6];  // W_b
    const float* r    = (const float*)d_in[7];  // r
    float* out = (float*)d_out;

    dim3 grid(B_DIM / 64, D_DIM / 64);   // 64 x 32 = 2048 blocks
    slstm_fused_kernel<<<grid, dim3(256), 0, stream>>>(x, c, n, h, m, W, bias, r, out);
}
